// LiquidNeuralNetwork_13924283973976
// MI455X (gfx1250) — compile-verified, matrix operand order checked by reading, not run on hardware
//
#include <hip/hip_runtime.h>
#include <hip/hip_bf16.h>
#include <math.h>

// ---------------------------------------------------------------------------
// Liquid RNN (2-layer, H=512, B=64, T=1024) for gfx1250.
// Persistent-kernel design: 32 WGs x 256 threads (8 wave32).
//   waves 0-3 : WMMA bf16 GEMM tiles (one 16x16 D tile per wave),
//               B-matrix (weights) served from LDS, A pipelined from global
//   waves 4-7 : decay (exp) precompute for the *other* layer, overlapped
// Device-wide sync: monotonic atomic-counter barrier (2 per timestep).
// ---------------------------------------------------------------------------

typedef __bf16 bf16_t;
typedef __attribute__((ext_vector_type(16))) __bf16 bf16x16;
typedef __attribute__((ext_vector_type(8)))  float  f32x8;

#define NWG 32
#define BLK 256

static constexpr int Bc = 64, Tt = 1024, DIN = 64, Hh = 512, DOUT = 3;
static constexpr int PAD = 16;                  // bf16 row pad: +32B -> 8-bank skew
static constexpr int LDH = Hh + PAD;            // padded row stride (big weights)
static constexpr int LDD = DIN + PAD;           // padded row stride (W0ih)

// ---- fragment load: 16x32 bf16 A-layout (B uses same layout on N-major W^T)
// lane L: m = L&15, half = L>>4
//   elems 0..7  <- K = half*8 + 0..7
//   elems 8..15 <- K = 16 + half*8 + 0..7
__device__ __forceinline__ bf16x16 load_frag(const bf16_t* tile, int ld) {
  const int lane = threadIdx.x & 31;
  const int m  = lane & 15;
  const int hf = lane >> 4;
  const bf16_t* p = tile + (size_t)m * ld + hf * 8;
  bf16x16 f;
#pragma unroll
  for (int j = 0; j < 8; ++j) f[j] = p[j];
#pragma unroll
  for (int j = 0; j < 8; ++j) f[8 + j] = p[16 + j];
  return f;
}

// ---- store 16x16 f32 accumulator as tanh(acc + b1[n] + b2[n]) in bf16 -----
// C/D layout: lane L: n = L&15; VGPR r -> m = 8*(L>>4) + r
__device__ __forceinline__ void store_tanh_tile(bf16_t* dst, int ld, f32x8 c,
                                                const float* bias1,
                                                const float* bias2,
                                                int row0, int n0) {
  const int lane = threadIdx.x & 31;
  const int n  = n0 + (lane & 15);
  const int hf = lane >> 4;
  const float bias = bias1[n] + bias2[n];
#pragma unroll
  for (int r = 0; r < 8; ++r) {
    int m = row0 + hf * 8 + r;
    dst[(size_t)m * ld + n] = (bf16_t)tanhf(c[r] + bias);
  }
}

// ---- device-wide barrier: monotonic counter, no reset needed --------------
__device__ __forceinline__ void grid_barrier(unsigned* cnt) {
  __syncthreads();
  if (threadIdx.x == 0) {
    __threadfence();
    unsigned a = __hip_atomic_fetch_add(cnt, 1u, __ATOMIC_ACQ_REL,
                                        __HIP_MEMORY_SCOPE_AGENT);
    unsigned need = (a / NWG + 1u) * NWG;
    while (__hip_atomic_load(cnt, __ATOMIC_ACQUIRE,
                             __HIP_MEMORY_SCOPE_AGENT) < need)
      __builtin_amdgcn_s_sleep(2);
  }
  __syncthreads();
}

// ---- stage one weight slice (16 rows x K) global -> padded LDS ------------
__device__ __forceinline__ void stage_weight(const bf16_t* __restrict__ g,
                                             bf16_t* __restrict__ l,
                                             int K, int ldl) {
  const int cpr = K / 8;              // 16B chunks per row
  const int total = 16 * cpr;
  for (int i = threadIdx.x; i < total; i += BLK) {
    int r = i / cpr, cch = i % cpr;
    const uint4* src = (const uint4*)(g + (size_t)r * K) + cch;
    uint4* dst = (uint4*)(l + (size_t)r * ldl) + cch;
    *dst = *src;
  }
}

// ---------------------------------------------------------------------------
// Setup kernels (re-run every call: deterministic, graph-capture safe)
// ---------------------------------------------------------------------------

// input_sequence [B,T,DIN] f32 -> Xbf [T,B,DIN] bf16 (time-major)
__global__ void pack_x_kernel(const float* __restrict__ x,
                              bf16_t* __restrict__ X) {
  int i = blockIdx.x * blockDim.x + threadIdx.x;
  if (i >= Tt * Bc * DIN) return;
  int t = i / (Bc * DIN);
  int r = i % (Bc * DIN);
  int b = r / DIN;
  int k = r % DIN;
  X[i] = (bf16_t)x[(size_t)b * Tt * DIN + (size_t)t * DIN + k];
}

// W [K,N] f32 -> Wt [N,K] bf16 (N-major so B-fragments load like A-fragments)
__global__ void pack_wt_kernel(const float* __restrict__ W,
                               bf16_t* __restrict__ Wt, int K, int N) {
  int i = blockIdx.x * blockDim.x + threadIdx.x;
  if (i >= K * N) return;
  int n = i / K;
  int k = i % K;
  Wt[i] = (bf16_t)W[(size_t)k * N + n];
}

// ---------------------------------------------------------------------------
// Persistent recurrent kernel
// ---------------------------------------------------------------------------
__global__ void __launch_bounds__(BLK)
lnn_persistent(const float* __restrict__ dt,       // [B,T]
               const float* __restrict__ b_ih0, const float* __restrict__ b_hh0,
               const float* __restrict__ w_td0, const float* __restrict__ b_td0,
               const float* __restrict__ b_ih1, const float* __restrict__ b_hh1,
               const float* __restrict__ w_td1, const float* __restrict__ b_td1,
               const float* __restrict__ w_out, const float* __restrict__ b_out,
               const bf16_t* __restrict__ Xbf,    // [T,B,DIN]
               const bf16_t* __restrict__ W0ih,   // [H,DIN]  N-major
               const bf16_t* __restrict__ W0hh,   // [H,H]    N-major
               const bf16_t* __restrict__ W1ih,   // [H,H]    N-major
               const bf16_t* __restrict__ W1hh,   // [H,H]    N-major
               bf16_t* __restrict__ h0, bf16_t* __restrict__ h1,
               bf16_t* __restrict__ hd0, bf16_t* __restrict__ hd1,
               unsigned* __restrict__ barcnt,
               float* __restrict__ out) {
  const int g    = blockIdx.x;        // 0..31 : owns N cols [16g, 16g+16)
  const int tid  = threadIdx.x;
  const int wave = tid >> 5;
  const int n0   = g * 16;

  // ---- LDS: this WG's 16-column weight slices, row-padded vs bank aliasing
  __shared__ bf16_t lW0ih[16 * LDD];   //  2.5 KB
  __shared__ bf16_t lW0hh[16 * LDH];   // 16.5 KB
  __shared__ bf16_t lW1ih[16 * LDH];   // 16.5 KB
  __shared__ bf16_t lW1hh[16 * LDH];   // 16.5 KB

  stage_weight(W0ih + (size_t)n0 * DIN, lW0ih, DIN, LDD);
  stage_weight(W0hh + (size_t)n0 * Hh,  lW0hh, Hh,  LDH);
  stage_weight(W1ih + (size_t)n0 * Hh,  lW1ih, Hh,  LDH);
  stage_weight(W1hh + (size_t)n0 * Hh,  lW1hh, Hh,  LDH);
  __syncthreads();

  for (int t = 0; t < Tt; ++t) {
    // ================= Phase A: layer-0 GEMM  ||  hd1 decay ================
    if (wave < 4) {
      const int row0 = wave * 16;
      f32x8 c = {};
      // x_t @ W_ih0  (K = 64), B from LDS
      const bf16_t* xt = Xbf + (size_t)t * Bc * DIN + (size_t)row0 * DIN;
#pragma unroll
      for (int kk = 0; kk < DIN; kk += 32) {
        bf16x16 a = load_frag(xt + kk, DIN);
        bf16x16 b = load_frag(lW0ih + kk, LDD);
        c = __builtin_amdgcn_wmma_f32_16x16x32_bf16(false, a, false, b,
                                                    (short)0, c, false, false);
      }
      // (h0*decay0) @ W_hh0  (K = 512), A pipelined, B from LDS
      const bf16_t* arow = hd0 + (size_t)row0 * Hh;
      bf16x16 a = load_frag(arow, Hh);
#pragma unroll
      for (int kk = 0; kk < Hh; kk += 32) {
        bf16x16 an = a;
        if (kk + 32 < Hh) an = load_frag(arow + kk + 32, Hh);
        bf16x16 b = load_frag(lW0hh + kk, LDH);
        c = __builtin_amdgcn_wmma_f32_16x16x32_bf16(false, a, false, b,
                                                    (short)0, c, false, false);
        a = an;
      }
      store_tanh_tile(h0, Hh, c, b_ih0, b_hh0, row0, n0);
    } else {
      // hd1 = h1 * exp(-|dt[:,t]*w_td1 + b_td1|)   (this step's layer-1 decay)
      const int lt   = tid - 128;                    // 0..127
      const int base = g * (Bc * Hh / NWG) + lt * 8; // 8 elems / thread
#pragma unroll
      for (int j = 0; j < 8; ++j) {
        int i = base + j;
        int b = i >> 9;      // / 512
        int k = i & 511;
        float d = expf(-fabsf(dt[(size_t)b * Tt + t] * w_td1[k] + b_td1[k]));
        hd1[i] = (bf16_t)((float)h1[i] * d);
      }
    }
    grid_barrier(barcnt);

    // ============ Phase B: layer-1 GEMM  ||  hd0 decay for t+1 =============
    if (wave < 4) {
      const int row0 = wave * 16;
      f32x8 c = {};
      const bf16_t* a1row = h0  + (size_t)row0 * Hh;
      const bf16_t* a2row = hd1 + (size_t)row0 * Hh;
      bf16x16 a1 = load_frag(a1row, Hh);
      bf16x16 a2 = load_frag(a2row, Hh);
#pragma unroll
      for (int kk = 0; kk < Hh; kk += 32) {
        bf16x16 a1n = a1, a2n = a2;
        if (kk + 32 < Hh) {
          a1n = load_frag(a1row + kk + 32, Hh);
          a2n = load_frag(a2row + kk + 32, Hh);
        }
        bf16x16 b1 = load_frag(lW1ih + kk, LDH);
        c = __builtin_amdgcn_wmma_f32_16x16x32_bf16(false, a1, false, b1,
                                                    (short)0, c, false, false);
        bf16x16 b2 = load_frag(lW1hh + kk, LDH);
        c = __builtin_amdgcn_wmma_f32_16x16x32_bf16(false, a2, false, b2,
                                                    (short)0, c, false, false);
        a1 = a1n; a2 = a2n;
      }
      store_tanh_tile(h1, Hh, c, b_ih1, b_hh1, row0, n0);
    } else {
      // hd0 for step t+1 = h0_new * exp(-|dt[:,t+1]*w_td0 + b_td0|)
      const int tn   = (t + 1 < Tt) ? (t + 1) : t;
      const int lt   = tid - 128;
      const int base = g * (Bc * Hh / NWG) + lt * 8;
#pragma unroll
      for (int j = 0; j < 8; ++j) {
        int i = base + j;
        int b = i >> 9;
        int k = i & 511;
        float d = expf(-fabsf(dt[(size_t)b * Tt + tn] * w_td0[k] + b_td0[k]));
        hd0[i] = (bf16_t)((float)h0[i] * d);
      }
    }
    grid_barrier(barcnt);
  }

  // ---- output projection: out = h1 @ w_out + b_out  ([64,512]@[512,3]) ----
  if (g == 0) {
    for (int i = tid; i < Bc * DOUT; i += BLK) {
      int b = i / DOUT, o = i % DOUT;
      float s = b_out[o];
      for (int k = 0; k < Hh; ++k)
        s += (float)h1[(size_t)b * Hh + k] * w_out[(size_t)k * DOUT + o];
      out[i] = s;
    }
  }
}

// ---------------------------------------------------------------------------
// Launch
// ---------------------------------------------------------------------------
extern "C" void kernel_launch(void* const* d_in, const int* in_sizes, int n_in,
                              void* d_out, int out_size, void* d_ws,
                              size_t ws_size, hipStream_t stream) {
  const float* x     = (const float*)d_in[0];   // [B,T,DIN]
  const float* dt    = (const float*)d_in[1];   // [B,T]
  const float* w_ih0 = (const float*)d_in[2];   // [DIN,H]
  const float* b_ih0 = (const float*)d_in[3];
  const float* w_hh0 = (const float*)d_in[4];   // [H,H]
  const float* b_hh0 = (const float*)d_in[5];
  const float* w_td0 = (const float*)d_in[6];
  const float* b_td0 = (const float*)d_in[7];
  const float* w_ih1 = (const float*)d_in[8];   // [H,H]
  const float* b_ih1 = (const float*)d_in[9];
  const float* w_hh1 = (const float*)d_in[10];  // [H,H]
  const float* b_hh1 = (const float*)d_in[11];
  const float* w_td1 = (const float*)d_in[12];
  const float* b_td1 = (const float*)d_in[13];
  const float* w_out = (const float*)d_in[14];  // [H,DOUT]
  const float* b_out = (const float*)d_in[15];
  float* out = (float*)d_out;

  // ---- workspace carve-out (256B aligned) ----
  char* ws = (char*)d_ws;
  size_t off = 0;
  auto carve = [&](size_t bytes) -> void* {
    void* p = ws + off;
    off = (off + bytes + 255) & ~(size_t)255;
    return p;
  };
  bf16_t* Xbf  = (bf16_t*)carve((size_t)Tt * Bc * DIN * 2);  // 8 MB
  bf16_t* W0ih = (bf16_t*)carve((size_t)Hh * DIN * 2);
  bf16_t* W0hh = (bf16_t*)carve((size_t)Hh * Hh * 2);
  bf16_t* W1ih = (bf16_t*)carve((size_t)Hh * Hh * 2);
  bf16_t* W1hh = (bf16_t*)carve((size_t)Hh * Hh * 2);
  bf16_t* h0   = (bf16_t*)carve((size_t)Bc * Hh * 2);
  bf16_t* h1   = (bf16_t*)carve((size_t)Bc * Hh * 2);
  bf16_t* hd0  = (bf16_t*)carve((size_t)Bc * Hh * 2);
  bf16_t* hd1  = (bf16_t*)carve((size_t)Bc * Hh * 2);
  unsigned* barcnt = (unsigned*)carve(256);

  // zero recurrent state + barrier counter every call (deterministic)
  hipMemsetAsync(h0, 0, (size_t)Bc * Hh * 2, stream);
  hipMemsetAsync(h1, 0, (size_t)Bc * Hh * 2, stream);
  hipMemsetAsync(hd0, 0, (size_t)Bc * Hh * 2, stream);
  hipMemsetAsync(hd1, 0, (size_t)Bc * Hh * 2, stream);
  hipMemsetAsync(barcnt, 0, 256, stream);

  // pack activations + weights to bf16 (weights N-major)
  {
    int n = Tt * Bc * DIN;
    pack_x_kernel<<<(n + 255) / 256, 256, 0, stream>>>(x, Xbf);
  }
  {
    int n = DIN * Hh;
    pack_wt_kernel<<<(n + 255) / 256, 256, 0, stream>>>(w_ih0, W0ih, DIN, Hh);
  }
  {
    int n = Hh * Hh;
    pack_wt_kernel<<<(n + 255) / 256, 256, 0, stream>>>(w_hh0, W0hh, Hh, Hh);
    pack_wt_kernel<<<(n + 255) / 256, 256, 0, stream>>>(w_ih1, W1ih, Hh, Hh);
    pack_wt_kernel<<<(n + 255) / 256, 256, 0, stream>>>(w_hh1, W1hh, Hh, Hh);
  }

  // persistent recurrent kernel: 32 WGs, all co-resident
  lnn_persistent<<<NWG, BLK, 0, stream>>>(
      dt, b_ih0, b_hh0, w_td0, b_td0, b_ih1, b_hh1, w_td1, b_td1,
      w_out, b_out, Xbf, W0ih, W0hh, W1ih, W1hh, h0, h1, hd0, hd1,
      barcnt, out);
}
